// HBMP_30803505447359
// MI455X (gfx1250) — matrix operand order repaired and checked by reading, not compiled, not run on hardware
//
#include <hip/hip_runtime.h>
#include <stdint.h>

#define TT 512
#define BB 64
#define EE 300
#define EP 320
#define HH 512
#define G4 2048
#define NWG 16
#define OUTC 3072

typedef float v8f __attribute__((ext_vector_type(8)));
typedef __bf16 v16bf __attribute__((ext_vector_type(16)));
typedef unsigned u32x4 __attribute__((ext_vector_type(4)));
typedef int i32x8 __attribute__((ext_vector_type(8)));
typedef int i32x4 __attribute__((ext_vector_type(4)));

union U32B { v16bf v; uint4 q[2]; };

__device__ __forceinline__ unsigned short f2bf(float f) {
    unsigned u = __float_as_uint(f);
    u += 0x7fffu + ((u >> 16) & 1u);
    return (unsigned short)(u >> 16);
}
__device__ __forceinline__ float sigf(float x) {
    return 1.0f / (1.0f + __builtin_amdgcn_exp2f(-1.442695040888963f * x));
}
__device__ __forceinline__ float tanhfast(float x) {
    return 2.0f / (1.0f + __builtin_amdgcn_exp2f(-2.885390081777927f * x)) - 1.0f;
}

// ---- Tensor Data Mover: 2D tile (tile_x elems of 2B) x tile_y rows ----------
// Global row stride = stride_x elems. Optional LDS padding: pad_amt_code+1
// DWORDs inserted every (2<<pad_int_code) DWORDs (per CDNA5 D# spec, §8.4).
// Toolchain uses the 6-arg builtin: (g0, g1, g2, g3, g4, cpol); trailing
// groups are unused for a 2D non-gather/non-iterate tile -> zero fill.
__device__ __forceinline__ void tdm_load_2d(unsigned lds_off, const void* gptr,
                                            unsigned tile_x, unsigned tile_y,
                                            unsigned stride_x, unsigned pad_en,
                                            unsigned pad_int_code, unsigned pad_amt_code)
{
    unsigned long long ga = (unsigned long long)gptr;
    u32x4 g0;
    g0[0] = 1u;                                          // count=1 (valid), user D#
    g0[1] = lds_off;                                     // lds_addr (bytes)
    g0[2] = (unsigned)(ga & 0xffffffffu);                // global_addr[31:0]
    g0[3] = (unsigned)((ga >> 32) & 0x1ffffffu) | (2u << 30); // addr[56:32], type=2
    i32x8 g1;
    g1[0] = (int)((1u << 16) |                           // data_size = 2 bytes
                  (pad_en << 20) | (pad_int_code << 22) | (pad_amt_code << 25));
    g1[1] = (int)((tile_x & 0xffffu) << 16);             // tensor_dim0[15:0]
    g1[2] = (int)(((tile_x >> 16) & 0xffffu) | ((tile_y & 0xffffu) << 16)); // dim0 hi, dim1 lo
    g1[3] = (int)(((tile_y >> 16) & 0xffffu) | ((tile_x & 0xffffu) << 16)); // dim1 hi, tile_dim0
    g1[4] = (int)(tile_y & 0xffffu);                     // tile_dim1, tile_dim2=0
    g1[5] = (int)stride_x;                               // tensor_dim0_stride[31:0]
    g1[6] = 0;                                           // stride0 hi, stride1 lo
    g1[7] = 0;                                           // stride1 hi
    i32x4 gz4 = {0, 0, 0, 0};
    i32x8 gz8 = {0, 0, 0, 0, 0, 0, 0, 0};
    __builtin_amdgcn_tensor_load_to_lds(g0, g1, gz4, gz4, gz8, 0);
}

// ---------------- pack x: [B][T][E] f32 -> [T][B][EP] bf16 (zero-padded K) ----
__global__ void pack_x_kernel(const float* __restrict__ x, unsigned short* __restrict__ xp) {
    size_t idx = (size_t)blockIdx.x * blockDim.x + threadIdx.x;
    if (idx >= (size_t)TT * BB * EP) return;
    int e  = (int)(idx % EP);
    int tb = (int)(idx / EP);
    int b  = tb % BB;
    int t  = tb / BB;
    float v = 0.0f;
    if (e < EE) v = x[((size_t)b * TT + t) * EE + e];
    xp[idx] = f2bf(v);
}

// ---------------- pack W: [K][4H] f32 -> W^T [4H][Kp] bf16 (zero-padded K) ----
__global__ void pack_wT_kernel(const float* __restrict__ w, unsigned short* __restrict__ wt,
                               int K, int Kp) {
    size_t idx = (size_t)blockIdx.x * blockDim.x + threadIdx.x;
    if (idx >= (size_t)G4 * Kp) return;
    int k = (int)(idx % Kp);
    int n = (int)(idx / Kp);
    float v = (k < K) ? w[(size_t)k * G4 + n] : 0.0f;
    wt[idx] = f2bf(v);
}

// ---------------- WMMA GEMM: out[r][c] = sum_k A[r][k]*Wt[c][k] + bias[c] ----
// A: [rows][Kp] bf16 row-major, Wt: [2048][Kp] bf16 (pre-transposed weights).
// Block tile 128x128, 8 waves, wave = 2 M-tiles x 4 N-tiles, k-step 32.
// Staging is software-pipelined: next k-tile is loaded to VGPRs while the
// current tile's WMMAs run, so s_wait_loadcnt overlaps with compute.
__global__ __launch_bounds__(256) void gemm_bf16_kernel(
    const unsigned short* __restrict__ A, int Kp,
    const unsigned short* __restrict__ Wt,
    const float* __restrict__ bias,
    float* __restrict__ out)
{
    __shared__ unsigned short As[128 * 40]; // padded stride 40 to dodge bank conflicts
    __shared__ unsigned short Bs[128 * 40];
    const int tid  = threadIdx.x;
    const int lane = tid & 31;
    const int wv   = tid >> 5;
    const int hi   = (lane >> 4) & 1;
    const int l16  = lane & 15;
    const size_t rb = (size_t)blockIdx.x * 128;
    const size_t cb = (size_t)blockIdx.y * 128;

    v8f acc[2][4];
    for (int i = 0; i < 2; ++i)
        for (int j = 0; j < 4; ++j)
            for (int v = 0; v < 8; ++v) acc[i][j][v] = 0.0f;

    // each thread owns 2 A-chunks and 2 B-chunks (16B each) per k-tile
    int rowc[2], k8c[2];
#pragma unroll
    for (int i = 0; i < 2; ++i) {
        int c = tid + 256 * i;
        rowc[i] = c >> 2;
        k8c[i]  = (c & 3) * 8;
    }
    uint4 ra[2], rbv[2];
    const int kIters = Kp >> 5;
    // prologue: load k-tile 0 into registers
#pragma unroll
    for (int i = 0; i < 2; ++i) {
        ra[i]  = *(const uint4*)(A  + (rb + rowc[i]) * Kp + k8c[i]);
        rbv[i] = *(const uint4*)(Wt + (cb + rowc[i]) * Kp + k8c[i]);
    }
    for (int kt = 0; kt < kIters; ++kt) {
        __syncthreads();   // previous compute finished reading LDS
#pragma unroll
        for (int i = 0; i < 2; ++i) {
            *(uint4*)(As + rowc[i] * 40 + k8c[i]) = ra[i];
            *(uint4*)(Bs + rowc[i] * 40 + k8c[i]) = rbv[i];
        }
        __syncthreads();
        if (kt + 1 < kIters) {   // issue next tile's loads; waited next iter
            int ko = (kt + 1) * 32;
#pragma unroll
            for (int i = 0; i < 2; ++i) {
                ra[i]  = *(const uint4*)(A  + (rb + rowc[i]) * Kp + ko + k8c[i]);
                rbv[i] = *(const uint4*)(Wt + (cb + rowc[i]) * Kp + ko + k8c[i]);
            }
        }
        U32B a[2], b[4];
#pragma unroll
        for (int i = 0; i < 2; ++i) {       // A 16x32: lane row=M, K chunks hi*8, hi*8+16
            int row = ((wv & 3) * 2 + i) * 16 + l16;
            a[i].q[0] = *(const uint4*)(As + row * 40 + hi * 8);
            a[i].q[1] = *(const uint4*)(As + row * 40 + hi * 8 + 16);
        }
#pragma unroll
        for (int j = 0; j < 4; ++j) {       // B 32x16: lane col=N, K chunks hi*8, hi*8+16
            int row = ((wv >> 2) * 4 + j) * 16 + l16;
            b[j].q[0] = *(const uint4*)(Bs + row * 40 + hi * 8);
            b[j].q[1] = *(const uint4*)(Bs + row * 40 + hi * 8 + 16);
        }
#pragma unroll
        for (int i = 0; i < 2; ++i)
#pragma unroll
            for (int j = 0; j < 4; ++j)
                acc[i][j] = __builtin_amdgcn_wmma_f32_16x16x32_bf16(
                    false, a[i].v, false, b[j].v, (short)0, acc[i][j], false, false);
    }
#pragma unroll
    for (int i = 0; i < 2; ++i)
#pragma unroll
        for (int j = 0; j < 4; ++j)
#pragma unroll
            for (int v = 0; v < 8; ++v) {
                size_t row = rb + ((wv & 3) * 2 + i) * 16 + v + 8 * hi;
                size_t col = cb + ((wv >> 2) * 4 + j) * 16 + l16;
                out[row * G4 + col] = acc[i][j][v] + bias[col];
            }
}

// ---------------- persistent LSTM recurrence ---------------------------------
// 16 WGs; WG k owns hidden units [32k,32k+32): Wh^T slab [128][512] bf16 in LDS,
// loaded once via TDM (4 gate blocks). Each step the 64x512 bf16 h_{t-1} block
// is DMA'd into LDS by the Tensor Data Mover with HW row padding (4 DWORDs per
// 256 DWORDs -> 520-element padded stride), freeing VALU/VMEM on the critical
// path. mode 0: store h-seq. mode 1: fwd+maxpool. mode 2: bwd (reversed xz).
__global__ __launch_bounds__(256) void lstm_recur_kernel(
    const float* __restrict__ xz,          // [T][B][2048] f32
    const unsigned short* __restrict__ WhT,// [2048][512] bf16
    unsigned short* __restrict__ h1,       // [T][B][512] bf16 (mode 0)
    unsigned short* __restrict__ hbuf,     // [2][B][512] bf16 (modes 1,2)
    const unsigned short* __restrict__ zblk,// [B][512] bf16 zeros
    float* __restrict__ out, int outoff,   // modes 1,2
    unsigned* __restrict__ cnt, int mode)
{
    extern __shared__ char smem[];
    unsigned short* WhB = (unsigned short*)smem;   // 128 rows x 520 (pad)
    unsigned short* hA  = WhB + 128 * 520;         // 64 rows  x 520 (pad)
    float* zS = (float*)(hA + 64 * 520);           // 64 x 128
    float* cS = zS + 64 * 128;                     // 64 x 32
    float* mx = cS + 64 * 32;                      // 64 x 32

    const int tid  = threadIdx.x;
    const int lane = tid & 31;
    const int wv   = tid >> 5;
    const int hi   = (lane >> 4) & 1;
    const int l16  = lane & 15;
    const int wg   = blockIdx.x;

    const unsigned whb_off = (unsigned)(unsigned long long)(void*)WhB;
    const unsigned ha_off  = (unsigned)(unsigned long long)(void*)hA;

    if (wv == 0) {
        // Wh^T slab: 4 gate blocks of [32 rows x 512], padded LDS rows
#pragma unroll
        for (int g = 0; g < 4; ++g)
            tdm_load_2d(whb_off + (unsigned)g * 32u * 520u * 2u,
                        WhT + ((size_t)g * HH + (size_t)wg * 32) * HH,
                        HH, 32, HH, 1u, 7u, 3u);
    }
    for (int r = 0; r < 8; ++r) {
        int cell = r * 256 + tid;
        cS[cell] = 0.0f;
        mx[cell] = -3.0e38f;
    }
    if (wv == 0) __builtin_amdgcn_s_wait_tensorcnt(0);
    __syncthreads();

    for (int t = 0; t < TT; ++t) {
        const unsigned short* hp;
        if (mode == 0) hp = (t == 0) ? zblk : (h1 + (size_t)(t - 1) * BB * HH);
        else           hp = (t == 0) ? zblk : (hbuf + (size_t)((t - 1) & 1) * BB * HH);
        if (wv == 0) {
            tdm_load_2d(ha_off, hp, HH, BB, HH, 1u, 7u, 3u);
            __builtin_amdgcn_s_wait_tensorcnt(0);
        }
        __syncthreads();

        v8f acc[4];
        for (int j = 0; j < 4; ++j)
            for (int v = 0; v < 8; ++v) acc[j][v] = 0.0f;
        const int mrow = (wv & 3) * 16 + l16;
        for (int ks = 0; ks < 16; ++ks) {
            U32B a;
            a.q[0] = *(const uint4*)(hA + mrow * 520 + ks * 32 + hi * 8);
            a.q[1] = *(const uint4*)(hA + mrow * 520 + ks * 32 + hi * 8 + 16);
#pragma unroll
            for (int j = 0; j < 4; ++j) {
                int nrow = ((wv >> 2) * 4 + j) * 16 + l16;
                U32B b;
                b.q[0] = *(const uint4*)(WhB + nrow * 520 + ks * 32 + hi * 8);
                b.q[1] = *(const uint4*)(WhB + nrow * 520 + ks * 32 + hi * 8 + 16);
                acc[j] = __builtin_amdgcn_wmma_f32_16x16x32_bf16(
                    false, a.v, false, b.v, (short)0, acc[j], false, false);
            }
        }
#pragma unroll
        for (int j = 0; j < 4; ++j) {
            int ncol = ((wv >> 2) * 4 + j) * 16 + l16;
#pragma unroll
            for (int v = 0; v < 8; ++v) {
                int m = (wv & 3) * 16 + v + 8 * hi;
                zS[m * 128 + ncol] = acc[j][v];
            }
        }
        __syncthreads();

        const int xrow = (mode == 2) ? (TT - 1 - t) : t;
#pragma unroll
        for (int r = 0; r < 8; ++r) {
            int cell = r * 256 + tid;
            int bq = cell >> 5, j = cell & 31;
            size_t base = ((size_t)xrow * BB + bq) * G4 + wg * 32 + j;
            float zi = zS[bq * 128 + j]       + xz[base];
            float zf = zS[bq * 128 + 32 + j]  + xz[base + 512];
            float zg = zS[bq * 128 + 64 + j]  + xz[base + 1024];
            float zo = zS[bq * 128 + 96 + j]  + xz[base + 1536];
            float ig = sigf(zi), fg = sigf(zf), gg = tanhfast(zg), og = sigf(zo);
            float cv = fg * cS[cell] + ig * gg;
            cS[cell] = cv;
            float h = og * tanhfast(cv);
            if (mode == 0) {
                h1[((size_t)t * BB + bq) * HH + wg * 32 + j] = f2bf(h);
            } else {
                hbuf[(size_t)(t & 1) * BB * HH + (size_t)bq * HH + wg * 32 + j] = f2bf(h);
                float m0 = mx[cell];
                mx[cell] = (h > m0) ? h : m0;
            }
        }
        // device-wide step barrier: monotonically increasing counter
        __threadfence();
        __syncthreads();
        if (tid == 0) {
            __hip_atomic_fetch_add(cnt, 1u, __ATOMIC_ACQ_REL, __HIP_MEMORY_SCOPE_AGENT);
            unsigned target = (unsigned)(NWG * (t + 1));
            while (__hip_atomic_load(cnt, __ATOMIC_ACQUIRE, __HIP_MEMORY_SCOPE_AGENT) < target)
                __builtin_amdgcn_s_sleep(2);
        }
        __syncthreads();
    }

    if (mode != 0) {
        for (int r = 0; r < 8; ++r) {
            int cell = r * 256 + tid;
            int bq = cell >> 5, j = cell & 31;
            out[(size_t)bq * OUTC + outoff + wg * 32 + j] = mx[cell];
        }
    }
}

extern "C" void kernel_launch(void* const* d_in, const int* in_sizes, int n_in,
                              void* d_out, int out_size, void* d_ws, size_t ws_size,
                              hipStream_t stream)
{
    (void)in_sizes; (void)n_in; (void)out_size; (void)ws_size;
    const float* x   = (const float*)d_in[0];
    const float* uWx = (const float*)d_in[1];
    const float* uWh = (const float*)d_in[2];
    const float* ub  = (const float*)d_in[3];
    const float* fWx = (const float*)d_in[4];
    const float* fWh = (const float*)d_in[5];
    const float* fb  = (const float*)d_in[6];
    const float* bWx = (const float*)d_in[7];
    const float* bWh = (const float*)d_in[8];
    const float* bb  = (const float*)d_in[9];
    float* out = (float*)d_out;

    char* w = (char*)d_ws;
    auto carve = [&](size_t bytes) -> char* {
        char* p = w;
        w += (bytes + 255) & ~(size_t)255;
        return p;
    };
    unsigned short* XP  = (unsigned short*)carve((size_t)TT * BB * EP * 2); // x packed
    unsigned short* H1  = (unsigned short*)carve((size_t)TT * BB * HH * 2); // uni h-seq
    unsigned short* WXT = (unsigned short*)carve((size_t)G4 * HH * 2);      // Wx^T
    unsigned short* WHT = (unsigned short*)carve((size_t)G4 * HH * 2);      // Wh^T
    unsigned short* HB  = (unsigned short*)carve((size_t)2 * BB * HH * 2);  // h ping-pong
    unsigned short* ZB  = (unsigned short*)carve((size_t)BB * HH * 2);      // zero h0
    unsigned*       CNT = (unsigned*)carve(256);
    float*          XZ  = (float*)carve((size_t)TT * BB * G4 * 4);          // xz buffer

    (void)hipMemsetAsync(ZB, 0, (size_t)BB * HH * 2, stream);
    {
        size_t total = (size_t)TT * BB * EP;
        pack_x_kernel<<<(unsigned)((total + 255) / 256), 256, 0, stream>>>(x, XP);
    }

    const size_t smem = (size_t)(128 * 520 + 64 * 520) * 2 +
                        (size_t)(64 * 128 + 64 * 32 + 64 * 32) * 4; // 243 KB
    dim3 ggrid(TT * BB / 128, G4 / 128); // (256,16)

    for (int nb = 0; nb < 3; ++nb) {
        size_t wxoff = (size_t)nb * EE * G4;
        size_t whoff = (size_t)nb * HH * G4;
        size_t boff  = (size_t)nb * G4;
        // ---- unidirectional LSTM ----
        pack_wT_kernel<<<(G4 * EP) / 256, 256, 0, stream>>>(uWx + wxoff, WXT, EE, EP);
        pack_wT_kernel<<<(G4 * HH) / 256, 256, 0, stream>>>(uWh + whoff, WHT, HH, HH);
        gemm_bf16_kernel<<<ggrid, 256, 0, stream>>>(XP, EP, WXT, ub + boff, XZ);
        (void)hipMemsetAsync(CNT, 0, 4, stream);
        lstm_recur_kernel<<<NWG, 256, smem, stream>>>(XZ, WHT, H1, HB, ZB,
                                                      nullptr, 0, CNT, 0);
        // ---- forward LSTM + maxpool ----
        pack_wT_kernel<<<(G4 * HH) / 256, 256, 0, stream>>>(fWx + whoff, WXT, HH, HH);
        pack_wT_kernel<<<(G4 * HH) / 256, 256, 0, stream>>>(fWh + whoff, WHT, HH, HH);
        gemm_bf16_kernel<<<ggrid, 256, 0, stream>>>(H1, HH, WXT, fb + boff, XZ);
        (void)hipMemsetAsync(CNT, 0, 4, stream);
        lstm_recur_kernel<<<NWG, 256, smem, stream>>>(XZ, WHT, nullptr, HB, ZB,
                                                      out, nb * 1024, CNT, 1);
        // ---- backward LSTM + maxpool ----
        pack_wT_kernel<<<(G4 * HH) / 256, 256, 0, stream>>>(bWx + whoff, WXT, HH, HH);
        pack_wT_kernel<<<(G4 * HH) / 256, 256, 0, stream>>>(bWh + whoff, WHT, HH, HH);
        gemm_bf16_kernel<<<ggrid, 256, 0, stream>>>(H1, HH, WXT, bb + boff, XZ);
        (void)hipMemsetAsync(CNT, 0, 4, stream);
        lstm_recur_kernel<<<NWG, 256, smem, stream>>>(XZ, WHT, nullptr, HB, ZB,
                                                      out, nb * 1024 + 512, CNT, 2);
    }
}
